// KLDivClustering_56092272886409
// MI455X (gfx1250) — matrix operand order repaired and checked by reading, not compiled, not run on hardware
//
#include <hip/hip_runtime.h>

// MI455X / gfx1250, wave32. D = A(bf16 16x32) x B(bf16 32x16) + C(f32) via
// v_wmma_f32_16x16x32_bf16. One wave owns a 16-row tile of N end-to-end.
// k-outer / ct-inner loop nest: each A chunk loaded once, 16 accumulators live.
// b1 staged via global_load_async_to_lds_b128 (ASYNCcnt); b2/b3 staged via the
// Tensor Data Mover (tensor_load_to_lds + s_wait_tensorcnt).

#define N_TOK 200000
#define D_IN  512
#define K_CL  256
#define XP    264   // padded bf16 row (256 + 8) -> 528B stride, conflict-free

typedef __attribute__((ext_vector_type(16))) __bf16 v16bf;
typedef __attribute__((ext_vector_type(8)))  __bf16 v8bf;
typedef __attribute__((ext_vector_type(8)))  float  v8f;
typedef __attribute__((ext_vector_type(4)))  float  v4f;
typedef __attribute__((ext_vector_type(4)))  unsigned int v4u;
typedef __attribute__((ext_vector_type(8)))  int    v8i;
typedef __attribute__((ext_vector_type(4)))  int    v4i;

static __device__ __forceinline__ v8f wmma_bf16(v16bf a, v16bf b, v8f c) {
    return __builtin_amdgcn_wmma_f32_16x16x32_bf16(false, a, false, b,
                                                   (short)0, c, false, false);
}

static __device__ __forceinline__ v16bf cat16(v8bf lo, v8bf hi) {
    return __builtin_shufflevector(lo, hi, 0,1,2,3,4,5,6,7,
                                   8,9,10,11,12,13,14,15);
}

#if __has_builtin(__builtin_amdgcn_tensor_load_to_lds)
#define HAVE_TDM 1
// 1-D contiguous TDM copy: nElems fp32 from gsrc -> LDS offset ldsOff.
// D# built from uniform values only (lands in SGPRs). Per CDNA5 ISA 8.3/8.4:
//   g0: [1:0]=count=1, [63:32]=lds_addr, [120:64]=global_addr, [127:126]=2
//   g1: [17:16]=data_size(2->4B), [79:48]=tensor_dim0, [127:112]=tile_dim0,
//       [207:160]=tensor_dim0_stride ; groups 2/3 zero (<=2-D tensor)
static __device__ __forceinline__ void tdm_load_1d(unsigned ldsOff,
                                                   const void* gsrc,
                                                   unsigned nElems) {
    unsigned long long ga = (unsigned long long)(size_t)gsrc;
    v4u g0;
    g0.x = 1u;                                                 // count=1
    g0.y = ldsOff;                                             // lds_addr
    g0.z = (unsigned)ga;                                       // gaddr[31:0]
    g0.w = ((unsigned)(ga >> 32) & 0x01FFFFFFu) | 0x80000000u; // [56:32]|type=2
    v8i g1;
    g1[0] = (int)(2u << 16);            // data_size = 4 bytes
    g1[1] = (int)(nElems << 16);        // tensor_dim0[15:0] @ bits 63:48
    g1[2] = (int)((nElems >> 16) | (1u << 16));  // dim0 hi | tensor_dim1 = 1
    g1[3] = (int)(nElems << 16);        // tile_dim0 @ bits 127:112
    g1[4] = 0;                          // tile_dim1/2 unused
    g1[5] = (int)nElems;                // tensor_dim0_stride[31:0]
    g1[6] = 0;
    g1[7] = 0;
    v4i gz = {0, 0, 0, 0};
#if defined(__clang_major__) && (__clang_major__ >= 23)
    v8i gz8 = {0, 0, 0, 0, 0, 0, 0, 0};
    __builtin_amdgcn_tensor_load_to_lds(g0, g1, gz, gz, gz8, 0);
#else
    __builtin_amdgcn_tensor_load_to_lds(g0, g1, gz, gz, 0);
#endif
}
#else
#define HAVE_TDM 0
#endif

// ---------------------------------------------------------------- prep ----
__global__ void prep_kernel(const float* __restrict__ W1,
                            const float* __restrict__ W2,
                            const float* __restrict__ W3,
                            __bf16* W1b, __bf16* W2b, __bf16* W3b,
                            float* gs, float* out_kl, float* out_mean) {
    int i = blockIdx.x * blockDim.x + threadIdx.x;
    if (i < K_CL * D_IN) { W1b[i] = (__bf16)W1[i]; out_mean[i] = 0.0f; }
    if (i < K_CL * K_CL) { W2b[i] = (__bf16)W2[i]; W3b[i] = (__bf16)W3[i]; }
    if (i < K_CL)        gs[i] = 0.0f;
    if (i == 0)          *out_kl = 0.0f;
}

// ----------------------------------------------------------------- mlp ----
__global__ __launch_bounds__(256) void
mlp_kernel(const float* __restrict__ emb,
           const __bf16* __restrict__ W1b,
           const __bf16* __restrict__ W2b,
           const __bf16* __restrict__ W3b,
           const float* __restrict__ b1,
           const float* __restrict__ b2,
           const float* __restrict__ b3,
           float* __restrict__ gammaOut,
           float* __restrict__ gsOut,
           float* __restrict__ idxOut) {
    extern __shared__ __bf16 lds[];
    const int wave = threadIdx.x >> 5;
    const int lane = threadIdx.x & 31;
    const int nn = lane & 15;                      // M (A) / N (B,C) index
    const int kh = lane >> 4;                      // lane-half select
    __bf16* x1 = lds + wave * (2 * 16 * XP);
    __bf16* x2 = x1 + 16 * XP;
    float*  biasLds = (float*)(lds + 8 * 2 * 16 * XP);   // 3*K_CL floats

    // ---- stage biases into LDS: b1 via async load, b2/b3 via TDM
    if (wave == 0) {
#pragma unroll
        for (int h = 0; h < 2; ++h) {            // b1: 2 x 512B async copies
            unsigned dst = (unsigned)(size_t)(void*)
                (biasLds + h * 128 + lane * 4);
            unsigned long long src = (unsigned long long)(size_t)
                (b1 + h * 128 + lane * 4);
            asm volatile("global_load_async_to_lds_b128 %0, %1, off"
                         :: "v"(dst), "v"(src) : "memory");
        }
#if HAVE_TDM
        tdm_load_1d((unsigned)(size_t)(void*)(biasLds + K_CL),     b2, K_CL);
        tdm_load_1d((unsigned)(size_t)(void*)(biasLds + 2 * K_CL), b3, K_CL);
#else
        const float* bsrc[2] = {b2, b3};
#pragma unroll
        for (int s = 0; s < 2; ++s) {
#pragma unroll
            for (int h = 0; h < 2; ++h) {
                unsigned dst = (unsigned)(size_t)(void*)
                    (biasLds + (s + 1) * K_CL + h * 128 + lane * 4);
                unsigned long long src = (unsigned long long)(size_t)
                    (bsrc[s] + h * 128 + lane * 4);
                asm volatile("global_load_async_to_lds_b128 %0, %1, off"
                             :: "v"(dst), "v"(src) : "memory");
            }
        }
#endif
        asm volatile("s_wait_asynccnt 0" ::: "memory");
#if HAVE_TDM
        __builtin_amdgcn_s_wait_tensorcnt(0);
#endif
    }
    __syncthreads();

    const int t = blockIdx.x * 8 + wave;           // wave-uniform
    if (t >= N_TOK / 16) return;                   // keeps EXEC all-ones
    const int n0 = t * 16;

    v8f acc[16];

    // ---- layer 1: x1 = leaky(emb @ W1^T + b1); A direct from global fp32,
    //      loaded ONCE per k-chunk and reused across all 16 output tiles.
#pragma unroll
    for (int ct = 0; ct < 16; ++ct) acc[ct] = (v8f){0,0,0,0,0,0,0,0};
    for (int kk = 0; kk < D_IN / 32; ++kk) {
        const float* Ap = emb + (size_t)(n0 + nn) * D_IN + kk * 32 + kh * 8;
        __builtin_prefetch(Ap + 32, 0, 0);         // global_prefetch_b8
        v4f f0 = *(const v4f*)(Ap);
        v4f f1 = *(const v4f*)(Ap + 4);
        v4f f2 = *(const v4f*)(Ap + 16);
        v4f f3 = *(const v4f*)(Ap + 20);
        v16bf a;
#pragma unroll
        for (int i = 0; i < 4; ++i) {
            a[i]      = (__bf16)f0[i];
            a[4 + i]  = (__bf16)f1[i];
            a[8 + i]  = (__bf16)f2[i];
            a[12 + i] = (__bf16)f3[i];
        }
#pragma unroll
        for (int ct = 0; ct < 16; ++ct) {
            const __bf16* bp = W1b + (size_t)(ct * 16 + nn) * D_IN
                             + kk * 32 + kh * 8;
            v16bf b = cat16(*(const v8bf*)bp, *(const v8bf*)(bp + 16));
            acc[ct] = wmma_bf16(a, b, acc[ct]);
        }
    }
#pragma unroll
    for (int ct = 0; ct < 16; ++ct) {
        const int col = ct * 16 + nn;
        const float bias = biasLds[col];
#pragma unroll
        for (int r = 0; r < 8; ++r) {
            float v = acc[ct][r] + bias;
            v = v > 0.0f ? v : 0.01f * v;
            x1[(kh * 8 + r) * XP + col] = (__bf16)v;
        }
    }

    // ---- layer 2: x2 = leaky(x1 @ W2^T + b2); A chunk from LDS once per kk
#pragma unroll
    for (int ct = 0; ct < 16; ++ct) acc[ct] = (v8f){0,0,0,0,0,0,0,0};
    for (int kk = 0; kk < K_CL / 32; ++kk) {
        const __bf16* Ap = x1 + nn * XP + kk * 32 + kh * 8;
        v16bf a = cat16(*(const v8bf*)Ap, *(const v8bf*)(Ap + 16));
#pragma unroll
        for (int ct = 0; ct < 16; ++ct) {
            const __bf16* bp = W2b + (size_t)(ct * 16 + nn) * K_CL
                             + kk * 32 + kh * 8;
            v16bf b = cat16(*(const v8bf*)bp, *(const v8bf*)(bp + 16));
            acc[ct] = wmma_bf16(a, b, acc[ct]);
        }
    }
#pragma unroll
    for (int ct = 0; ct < 16; ++ct) {
        const int col = ct * 16 + nn;
        const float bias = biasLds[K_CL + col];
#pragma unroll
        for (int r = 0; r < 8; ++r) {
            float v = acc[ct][r] + bias;
            v = v > 0.0f ? v : 0.01f * v;
            x2[(kh * 8 + r) * XP + col] = (__bf16)v;
        }
    }

    // ---- layer 3: logits kept in registers for fused softmax
#pragma unroll
    for (int ct = 0; ct < 16; ++ct) acc[ct] = (v8f){0,0,0,0,0,0,0,0};
    for (int kk = 0; kk < K_CL / 32; ++kk) {
        const __bf16* Ap = x2 + nn * XP + kk * 32 + kh * 8;
        v16bf a = cat16(*(const v8bf*)Ap, *(const v8bf*)(Ap + 16));
#pragma unroll
        for (int ct = 0; ct < 16; ++ct) {
            const __bf16* bp = W3b + (size_t)(ct * 16 + nn) * K_CL
                             + kk * 32 + kh * 8;
            v16bf b = cat16(*(const v8bf*)bp, *(const v8bf*)(bp + 16));
            acc[ct] = wmma_bf16(a, b, acc[ct]);
        }
    }
#pragma unroll
    for (int ct = 0; ct < 16; ++ct) {
        const float bias = biasLds[2 * K_CL + ct * 16 + nn];
#pragma unroll
        for (int r = 0; r < 8; ++r) acc[ct][r] += bias;
    }

    // ---- softmax + argmax + gamma store + gamma_sum partials
    float gspart[16];
#pragma unroll
    for (int ct = 0; ct < 16; ++ct) gspart[ct] = 0.0f;

    for (int r = 0; r < 8; ++r) {
        float mx = -3.4e38f; int mi = 0;
#pragma unroll
        for (int ct = 0; ct < 16; ++ct) {
            float v = acc[ct][r];
            int   c = ct * 16 + nn;
            if (v > mx) { mx = v; mi = c; }
        }
        for (int m = 1; m < 16; m <<= 1) {      // rows live in 16-lane halves
            float om = __shfl_xor(mx, m, 32);
            int   oi = __shfl_xor(mi, m, 32);
            if (om > mx || (om == mx && oi < mi)) { mx = om; mi = oi; }
        }
        float e[16]; float s = 0.0f;
#pragma unroll
        for (int ct = 0; ct < 16; ++ct) { e[ct] = __expf(acc[ct][r] - mx); s += e[ct]; }
        for (int m = 1; m < 16; m <<= 1) s += __shfl_xor(s, m, 32);
        const float inv = 1.0f / s;
        const int row = n0 + kh * 8 + r;
#pragma unroll
        for (int ct = 0; ct < 16; ++ct) {
            float g = e[ct] * inv;
            gspart[ct] += g;
            gammaOut[(size_t)row * K_CL + ct * 16 + nn] = g;
        }
        if (nn == 0) idxOut[row] = (float)mi;
    }
#pragma unroll
    for (int ct = 0; ct < 16; ++ct)
        atomicAdd(&gsOut[ct * 16 + nn], gspart[ct]);
}

// ------------------------------------------------------ mean = gamma^T@E ----
__global__ __launch_bounds__(256) void
meanacc_kernel(const float* __restrict__ emb,
               const float* __restrict__ gamma,
               float* __restrict__ meanOut) {
    const int wave = threadIdx.x >> 5;
    const int lane = threadIdx.x & 31;
    const int nn = lane & 15, kh = lane >> 4;
    const int m0 = blockIdx.x * 16;       // cluster tile (16 tiles)
    const int split = blockIdx.y;         // N split     (64 splits)
    v8f acc[4];
#pragma unroll
    for (int dt = 0; dt < 4; ++dt) acc[dt] = (v8f){0,0,0,0,0,0,0,0};

    for (int c = split; c < N_TOK / 32; c += 64) {
        const int nb = c * 32;
        // A = gamma^T 16x32 tile: M = cluster (nn), K = token rows
        v16bf a;
        const float* gp0 = gamma + (size_t)(nb + kh * 8) * K_CL + m0 + nn;
        const float* gp1 = gamma + (size_t)(nb + 16 + kh * 8) * K_CL + m0 + nn;
#pragma unroll
        for (int i = 0; i < 8; ++i) {
            a[i]     = (__bf16)gp0[(size_t)i * K_CL];
            a[8 + i] = (__bf16)gp1[(size_t)i * K_CL];
        }
#pragma unroll
        for (int dt = 0; dt < 4; ++dt) {
            const int d = (wave * 4 + dt) * 16 + nn;
            const float* ep = emb + (size_t)(nb + kh * 8) * D_IN + d;
            v16bf b;
#pragma unroll
            for (int i = 0; i < 16; ++i) b[i] = (__bf16)ep[(size_t)(i + (i >= 8 ? 8 : 0)) * D_IN];
            acc[dt] = wmma_bf16(a, b, acc[dt]);
        }
    }
#pragma unroll
    for (int dt = 0; dt < 4; ++dt) {
        const int d = (wave * 4 + dt) * 16 + nn;
#pragma unroll
        for (int r = 0; r < 8; ++r)
            atomicAdd(&meanOut[(size_t)(m0 + kh * 8 + r) * D_IN + d], acc[dt][r]);
    }
}

// ----------------------------------------------------------------- KL ----
__global__ __launch_bounds__(256) void
klred_kernel(const float* __restrict__ gamma,
             const float* __restrict__ gs,
             float* __restrict__ out_kl) {
    const int wave = threadIdx.x >> 5;
    const int lane = threadIdx.x & 31;
    const int row = blockIdx.x * 8 + wave;
    if (row >= N_TOK) return;
    const float* g = gamma + (size_t)row * K_CL + lane * 8;
    const float* q = gs + lane * 8;
    float gg[8], ss[8];
    float smax = -3.4e38f, acc = 0.0f;
#pragma unroll
    for (int j = 0; j < 8; ++j) {
        gg[j] = g[j];
        ss[j] = gg[j] * gg[j] / q[j];
        smax = fmaxf(smax, ss[j]);
    }
    for (int m = 1; m < 32; m <<= 1) smax = fmaxf(smax, __shfl_xor(smax, m, 32));
    float se = 0.0f;
#pragma unroll
    for (int j = 0; j < 8; ++j) {
        se  += __expf(ss[j] - smax);
        acc += (gg[j] > 0.0f ? gg[j] * __logf(gg[j]) : 0.0f) - gg[j] * ss[j];
    }
    for (int m = 1; m < 32; m <<= 1) {
        se  += __shfl_xor(se, m, 32);
        acc += __shfl_xor(acc, m, 32);
    }
    // sum_k gamma == 1 per row, so  kl_row = acc + lse
    if (lane == 0) atomicAdd(out_kl, acc + smax + __logf(se));
}

// ------------------------------------------------------------- finalize ----
__global__ void meandiv_kernel(float* mean, const float* __restrict__ gs) {
    int i = blockIdx.x * blockDim.x + threadIdx.x;
    if (i < K_CL * D_IN) mean[i] /= gs[i / D_IN];
}

// --------------------------------------------------------------- launch ----
extern "C" void kernel_launch(void* const* d_in, const int* in_sizes, int n_in,
                              void* d_out, int out_size, void* d_ws, size_t ws_size,
                              hipStream_t stream) {
    (void)in_sizes; (void)n_in; (void)out_size; (void)ws_size;
    const float* emb = (const float*)d_in[0];
    const float* W1  = (const float*)d_in[1];
    const float* b1  = (const float*)d_in[2];
    const float* W2  = (const float*)d_in[3];
    const float* b2  = (const float*)d_in[4];
    const float* W3  = (const float*)d_in[5];
    const float* b3  = (const float*)d_in[6];

    float* out      = (float*)d_out;
    float* out_kl   = out;                       // [1]
    float* out_idx  = out + 1;                   // [N]
    float* out_mean = out + 1 + N_TOK;           // [K*D]

    __bf16* W1b  = (__bf16*)d_ws;                // K*D bf16
    __bf16* W2b  = W1b + (size_t)K_CL * D_IN;    // K*K bf16
    __bf16* W3b  = W2b + (size_t)K_CL * K_CL;    // K*K bf16
    float*  gs   = (float*)(W3b + (size_t)K_CL * K_CL);  // [K]
    float*  gam  = gs + K_CL;                    // [N*K] fp32

    prep_kernel<<<(K_CL * D_IN + 255) / 256, 256, 0, stream>>>(
        W1, W2, W3, W1b, W2b, W3b, gs, out_kl, out_mean);

    const size_t shmem = (size_t)8 * 2 * 16 * XP * sizeof(__bf16)  // x buffers
                       + (size_t)3 * K_CL * sizeof(float);         // biases
    mlp_kernel<<<(N_TOK / 16 + 7) / 8, 256, shmem, stream>>>(
        emb, W1b, W2b, W3b, b1, b2, b3, gam, gs, out_idx);

    meanacc_kernel<<<dim3(16, 64), 256, 0, stream>>>(emb, gam, out_mean);

    klred_kernel<<<(N_TOK + 7) / 8, 256, 0, stream>>>(gam, gs, out_kl);

    meandiv_kernel<<<(K_CL * D_IN + 255) / 256, 256, 0, stream>>>(out_mean, gs);
}
